// Tactic_Selector_25134148616709
// MI455X (gfx1250) — compile-verified, matrix operand order checked by reading, not run on hardware
//
#include <hip/hip_runtime.h>
#include <hip/hip_bf16.h>

// ---------------------------------------------------------------------------
// Fully-fused Tactic-Selector cell for gfx1250 (MI455X).
//  * One workgroup (256 threads = 8 wave32) per batch element.
//  * All activations live in LDS as bf16; GEMMs via V_WMMA_F32_16X16X32_BF16.
//  * Weights are pre-converted to bf16 and pre-swizzled into WMMA B-fragment
//    order in d_ws by a one-shot kernel, so the hot loop does no conversion:
//    B-fragment = 2x global_load_b128, A-fragment = 2x ds_load_b128.
// ---------------------------------------------------------------------------

typedef __attribute__((ext_vector_type(16))) __bf16 v16bf;
typedef __attribute__((ext_vector_type(8)))  float  v8f;

#define BDIM 256
#define NAG  64
#define HD   256
#define DIN  256
#define TN   16

// --- float -> bf16 conversion: prefer the HW packed convert when available --
#if defined(__has_builtin)
#if __has_builtin(__builtin_amdgcn_cvt_pk_bf16_f32)
#define HAVE_CVT_PK_BF16 1
#endif
#endif

__device__ __forceinline__ unsigned pack2bf(float a, float b) {
#ifdef HAVE_CVT_PK_BF16
  auto r = __builtin_amdgcn_cvt_pk_bf16_f32(a, b);   // v_cvt_pk_bf16_f32
  return __builtin_bit_cast(unsigned, r);
#else
  unsigned ua = __builtin_bit_cast(unsigned, a);
  unsigned ub = __builtin_bit_cast(unsigned, b);
  ua += 0x7FFFu + ((ua >> 16) & 1u);                 // RNE
  ub += 0x7FFFu + ((ub >> 16) & 1u);
  return (ua >> 16) | (ub & 0xFFFF0000u);
#endif
}
__device__ __forceinline__ unsigned short f2bf(float f) {
  return (unsigned short)pack2bf(f, 0.0f);
}
__device__ __forceinline__ float bf2f(unsigned short s) {
  unsigned u = ((unsigned)s) << 16;
  return __builtin_bit_cast(float, u);
}

union FragB { v16bf v; unsigned u[8]; };

__device__ __forceinline__ v8f z8() {
  v8f z;
#pragma unroll
  for (int i = 0; i < 8; ++i) z[i] = 0.0f;
  return z;
}

// K offset inside a 32-wide K block for fragment dword v, lane half hi.
// Matches CDNA5 ISA 16-bit A/B layout (CDNA5 ISA §7.12.2).
__device__ __forceinline__ int kmap(int v, int hi) {
  return ((v & 3) * 2) + ((v >> 2) * 16) + hi * 8;
}

// A fragment (or row-major B fragment) from LDS bf16, leading dim ld.
// Per-lane data = two contiguous 16B runs -> 2x ds_load_b128.
__device__ __forceinline__ FragB frag_lds_rowmajor(const unsigned short* S, int ld,
                                                   int row0, int k0, int lane) {
  FragB f;
  int m = lane & 15, hi = lane >> 4;
  const unsigned short* p = S + (row0 + m) * ld + k0 + hi * 8;
  uint4 lo = *(const uint4*)p;
  uint4 hh = *(const uint4*)(p + 16);
  f.u[0] = lo.x; f.u[1] = lo.y; f.u[2] = lo.z; f.u[3] = lo.w;
  f.u[4] = hh.x; f.u[5] = hh.y; f.u[6] = hh.z; f.u[7] = hh.w;
  return f;
}

// B fragment from pre-swizzled bf16 weights in d_ws:
// layout [rowTile][kBlock][lane][8 dwords] -> 2x global_load_b128 per lane.
__device__ __forceinline__ FragB frag_w_sw(const unsigned* __restrict__ Wsw, int kbTot,
                                           int nt, int kb, int lane) {
  FragB f;
  const unsigned* p = Wsw + (((size_t)nt * kbTot + kb) * 32 + lane) * 8;
  uint4 lo = *(const uint4*)p;
  uint4 hh = *(const uint4*)(p + 4);
  f.u[0] = lo.x; f.u[1] = lo.y; f.u[2] = lo.z; f.u[3] = lo.w;
  f.u[4] = hh.x; f.u[5] = hh.y; f.u[6] = hh.z; f.u[7] = hh.w;
  return f;
}

__device__ __forceinline__ v8f wmma_bf16(const FragB& a, const FragB& b, v8f c) {
  return __builtin_amdgcn_wmma_f32_16x16x32_bf16(false, a.v, false, b.v,
                                                 (short)0, c, false, false);
}

// Wave-level GEMM: 16 rows x 128 cols; A in LDS bf16, W pre-swizzled in global.
// ntBase already includes the wave's column-half tile offset (c0/16).
__device__ __forceinline__ void gemm_sw(const unsigned short* A, int ldA, int r0,
                                        const unsigned* __restrict__ Wsw, int kbTot,
                                        int ntBase, int kbBase, int nkb,
                                        int lane, v8f acc[8]) {
  for (int kb = 0; kb < nkb; ++kb) {
    FragB a = frag_lds_rowmajor(A, ldA, r0, kb * 32, lane);
#pragma unroll
    for (int t = 0; t < 8; ++t) {
      FragB b = frag_w_sw(Wsw, kbTot, ntBase + t, kbBase + kb, lane);
      acc[t] = wmma_bf16(a, b, acc[t]);
    }
  }
}

__device__ __forceinline__ float act_apply(float x, int act) {
  if (act == 1) return x > 0.0f ? x : 0.0f;                  // relu
  if (act == 2) return 1.0f / (1.0f + __expf(-x));           // sigmoid
  return x;
}

// Store a wave's 16x128 accumulator region to LDS bf16 (+bias(+bias2), act).
__device__ __forceinline__ void store_lds(unsigned short* Out, int ldO, int r0, int c0,
                                          const float* b1, const float* b2, int boff,
                                          int act, v8f acc[8], int lane) {
  int n = lane & 15, hi = lane >> 4;
#pragma unroll
  for (int t = 0; t < 8; ++t) {
    int col = c0 + t * 16 + n;
    float bv = (b1 ? b1[boff + col] : 0.0f) + (b2 ? b2[boff + col] : 0.0f);
#pragma unroll
    for (int v = 0; v < 8; ++v) {
      int row = r0 + v + hi * 8;
      Out[row * ldO + col] = f2bf(act_apply(acc[t][v] + bv, act));
    }
  }
}

// Store transposed (Out[col][row], ld=64) -- used for V so attn@V B-fragments
// become row-major ds_load_b128 reads.
__device__ __forceinline__ void store_lds_t(unsigned short* Out, int r0, int c0,
                                            const float* b1, int boff,
                                            v8f acc[8], int lane) {
  int n = lane & 15, hi = lane >> 4;
#pragma unroll
  for (int t = 0; t < 8; ++t) {
    int col = c0 + t * 16 + n;
    float bv = b1[boff + col];
#pragma unroll
    for (int v = 0; v < 8; ++v) {
      int row = r0 + v + hi * 8;
      Out[col * NAG + row] = f2bf(acc[t][v] + bv);
    }
  }
}

// ---------------------------------------------------------------------------
// One-shot weight conversion: f32 row-major [OUT][K] -> bf16 swizzled
// [OUT/16][K/32][lane][8 dwords] matching the WMMA 16-bit B layout.
// ---------------------------------------------------------------------------
extern "C" __global__ void swizzle_w(const float* __restrict__ W, int OUT, int K,
                                     unsigned* __restrict__ dst) {
  int idx = blockIdx.x * BDIM + threadIdx.x;          // one thread per dst dword
  int total = OUT * (K >> 1);
  if (idx >= total) return;
  int v    = idx & 7;
  int lane = (idx >> 3) & 31;
  int blk  = idx >> 8;
  int kbTot = K >> 5;
  int kb = blk % kbTot;
  int nt = blk / kbTot;
  int n  = nt * 16 + (lane & 15);
  int k  = kb * 32 + kmap(v, lane >> 4);
  const float* p = W + (size_t)n * K + k;
  dst[idx] = pack2bf(p[0], p[1]);
}

// ---------------------------------------------------------------------------
// Main fused kernel.
// ---------------------------------------------------------------------------
extern "C" __global__ __launch_bounds__(BDIM) void tactic_fused(
    const float* __restrict__ inp, const float* __restrict__ hin,
    const unsigned char* __restrict__ drop,
    const unsigned* __restrict__ w_fc,   const float* __restrict__ fc_gru_b,
    const unsigned* __restrict__ w_inp,  const float* __restrict__ in_proj_b,
    const unsigned* __restrict__ w_outp, const float* __restrict__ out_proj_b,
    const unsigned* __restrict__ w_att,  const float* __restrict__ att_b,
    const unsigned* __restrict__ w_ih,   const unsigned* __restrict__ w_hh,
    const float* __restrict__ bih, const float* __restrict__ bhh,
    const unsigned* __restrict__ w_fc2, const float* __restrict__ fc2_b,
    float* __restrict__ out) {
  extern __shared__ unsigned short smem[];
  unsigned short* R0 = smem;               // staged inputs  -> z gate
  unsigned short* R1 = smem + 16384;       // x              -> h_n
  unsigned short* R2 = smem + 32768;       // h_in (bf16)
  unsigned short* R3 = smem + 49152;       // q -> ctx -> r gate
  unsigned short* R4 = smem + 65536;       // k -> x_att -> h_new (bf16)
  unsigned short* R5 = smem + 81920;       // Vt (transposed) -> x_
  float*          sS = (float*)(smem + 98304);   // scores 64x64 f32
  unsigned short* sP = smem + 106496;            // attn 64x64 bf16

  const int b    = blockIdx.x;
  const int tid  = threadIdx.x;
  const int wave = tid >> 5;
  const int lane = tid & 31;
  const int r0   = (wave >> 1) * 16;   // full-partition row block
  const int c0   = (wave & 1) * 128;   // full-partition column half
  const int ct0  = c0 >> 4;            // column tile base

  const float*         inB = inp + (size_t)b * NAG * DIN;
  const float*         hB  = hin + (size_t)b * NAG * HD;
  const unsigned char* dB  = drop + (size_t)b * NAG * NAG;

  const size_t TQ = (size_t)gridDim.x * NAG * TN;   // size of tactic_q region
  const size_t HT = (size_t)gridDim.x * NAG * HD;   // size of h region

  // ---- stage inputs / hidden into LDS as bf16 (float4 -> packed uint2) -----
  for (int i = tid; i < (NAG * DIN) / 4; i += BDIM) {
    float4 w = ((const float4*)inB)[i];
    uint2 p; p.x = pack2bf(w.x, w.y); p.y = pack2bf(w.z, w.w);
    ((uint2*)R0)[i] = p;
  }
  for (int i = tid; i < (NAG * HD) / 4; i += BDIM) {
    float4 w = ((const float4*)hB)[i];
    uint2 p; p.x = pack2bf(w.x, w.y); p.y = pack2bf(w.z, w.w);
    ((uint2*)R2)[i] = p;
  }
  __syncthreads();

  v8f acc[8];

  // ---- x = relu(inp @ fc_gru_w^T + b) -> R1 --------------------------------
#pragma unroll
  for (int t = 0; t < 8; ++t) acc[t] = z8();
  gemm_sw(R0, DIN, r0, w_fc, 8, ct0, 0, 8, lane, acc);
  store_lds(R1, HD, r0, c0, fc_gru_b, nullptr, 0, 1, acc, lane);
  __syncthreads();

  // ---- q,k,v = x @ in_proj^T + b -> R3, R4, R5(T) --------------------------
#pragma unroll
  for (int t = 0; t < 8; ++t) acc[t] = z8();
  gemm_sw(R1, HD, r0, w_inp, 8, ct0, 0, 8, lane, acc);
  store_lds(R3, HD, r0, c0, in_proj_b, nullptr, 0, 0, acc, lane);
#pragma unroll
  for (int t = 0; t < 8; ++t) acc[t] = z8();
  gemm_sw(R1, HD, r0, w_inp, 8, 16 + ct0, 0, 8, lane, acc);
  store_lds(R4, HD, r0, c0, in_proj_b, nullptr, 256, 0, acc, lane);
#pragma unroll
  for (int t = 0; t < 8; ++t) acc[t] = z8();
  gemm_sw(R1, HD, r0, w_inp, 8, 32 + ct0, 0, 8, lane, acc);
  store_lds_t(R5, r0, c0, in_proj_b, 512, acc, lane);   // V stored transposed
  __syncthreads();

  // ---- scores = (q k^T)/16, masked -> sS (each wave: 16 rows x 32 cols) ----
  {
    int sr = (wave & 3) * 16;
    int sc = (wave >> 2) * 32;
    v8f sa[2];
    sa[0] = z8(); sa[1] = z8();
    for (int k0 = 0; k0 < HD; k0 += 32) {
      FragB a = frag_lds_rowmajor(R3, HD, sr, k0, lane);
#pragma unroll
      for (int t = 0; t < 2; ++t) {
        FragB bb = frag_lds_rowmajor(R4, HD, sc + t * 16, k0, lane);
        sa[t] = wmma_bf16(a, bb, sa[t]);
      }
    }
    int n = lane & 15, hi = lane >> 4;
#pragma unroll
    for (int t = 0; t < 2; ++t) {
      int col = sc + t * 16 + n;
#pragma unroll
      for (int v = 0; v < 8; ++v) {
        int row = sr + v + hi * 8;
        float s = sa[t][v] * 0.0625f;
        if (dB[row * NAG + col]) s = -1.0e9f;
        sS[row * NAG + col] = s;
      }
    }
  }
  __syncthreads();

  // ---- softmax over rows (64 threads, one row each) -> sP (bf16) -----------
  if (tid < NAG) {
    float* rowp = sS + tid * NAG;
    float mx = rowp[0];
    for (int j = 1; j < NAG; ++j) mx = fmaxf(mx, rowp[j]);
    float sum = 0.0f;
    for (int j = 0; j < NAG; ++j) { float e = __expf(rowp[j] - mx); rowp[j] = e; sum += e; }
    float inv = 1.0f / sum;
    unsigned* prow = (unsigned*)(sP + tid * NAG);
    for (int j = 0; j < NAG / 2; ++j)
      prow[j] = pack2bf(rowp[2 * j] * inv, rowp[2 * j + 1] * inv);
  }
  __syncthreads();

  // ---- ctx = attn @ v -> R3 (reuse q); B from transposed V (row-major) -----
#pragma unroll
  for (int t = 0; t < 8; ++t) acc[t] = z8();
  for (int k0 = 0; k0 < NAG; k0 += 32) {
    FragB a = frag_lds_rowmajor(sP, NAG, r0, k0, lane);
#pragma unroll
    for (int t = 0; t < 8; ++t) {
      FragB bb = frag_lds_rowmajor(R5, NAG, c0 + t * 16, k0, lane);
      acc[t] = wmma_bf16(a, bb, acc[t]);
    }
  }
  store_lds(R3, HD, r0, c0, nullptr, nullptr, 0, 0, acc, lane);
  __syncthreads();

  // ---- x_att = ctx @ out_proj^T + b -> R4 ----------------------------------
#pragma unroll
  for (int t = 0; t < 8; ++t) acc[t] = z8();
  gemm_sw(R3, HD, r0, w_outp, 8, ct0, 0, 8, lane, acc);
  store_lds(R4, HD, r0, c0, out_proj_b, nullptr, 0, 0, acc, lane);
  __syncthreads();

  // ---- x_ = relu([x | x_att] @ att_w^T + b) -> R5 --------------------------
#pragma unroll
  for (int t = 0; t < 8; ++t) acc[t] = z8();
  gemm_sw(R1, HD, r0, w_att, 16, ct0, 0, 8, lane, acc);   // x part     (k-blocks 0..7)
  gemm_sw(R4, HD, r0, w_att, 16, ct0, 8, 8, lane, acc);   // x_att part (k-blocks 8..15)
  store_lds(R5, HD, r0, c0, att_b, nullptr, 0, 1, acc, lane);
  __syncthreads();

  // ---- GRU gates: r -> R3, z -> R0 -----------------------------------------
#pragma unroll
  for (int t = 0; t < 8; ++t) acc[t] = z8();
  gemm_sw(R5, HD, r0, w_ih, 8, ct0, 0, 8, lane, acc);
  gemm_sw(R2, HD, r0, w_hh, 8, ct0, 0, 8, lane, acc);
  store_lds(R3, HD, r0, c0, bih, bhh, 0, 2, acc, lane);
#pragma unroll
  for (int t = 0; t < 8; ++t) acc[t] = z8();
  gemm_sw(R5, HD, r0, w_ih, 8, 16 + ct0, 0, 8, lane, acc);
  gemm_sw(R2, HD, r0, w_hh, 8, 16 + ct0, 0, 8, lane, acc);
  store_lds(R0, HD, r0, c0, bih, bhh, 256, 2, acc, lane);
  __syncthreads();

  // ---- h_n = h_in @ whh_n^T + bhh_n -> R1 (bf16 staging) -------------------
#pragma unroll
  for (int t = 0; t < 8; ++t) acc[t] = z8();
  gemm_sw(R2, HD, r0, w_hh, 8, 32 + ct0, 0, 8, lane, acc);
  store_lds(R1, HD, r0, c0, bhh, nullptr, 512, 0, acc, lane);
  __syncthreads();

  // ---- i_n + combine: n = tanh(i_n + r*h_n); h = (1-z)*n + z*h_in ----------
#pragma unroll
  for (int t = 0; t < 8; ++t) acc[t] = z8();
  gemm_sw(R5, HD, r0, w_ih, 8, 32 + ct0, 0, 8, lane, acc);
  {
    int n = lane & 15, hi = lane >> 4;
    float* hOut = out + TQ + (size_t)b * NAG * HD;
#pragma unroll
    for (int t = 0; t < 8; ++t) {
      int col = c0 + t * 16 + n;
      float bin = bih[512 + col];
#pragma unroll
      for (int v = 0; v < 8; ++v) {
        int row = r0 + v + hi * 8;
        float i_n = acc[t][v] + bin;
        float h_n = bf2f(R1[row * HD + col]);
        float rr  = bf2f(R3[row * HD + col]);
        float zz  = bf2f(R0[row * HD + col]);
        float nn  = tanhf(i_n + rr * h_n);
        float hp  = hB[row * HD + col];          // exact f32 previous hidden
        float hn  = (1.0f - zz) * nn + zz * hp;
        hOut[row * HD + col] = hn;
        R4[row * HD + col] = f2bf(hn);
      }
    }
  }
  __syncthreads();

  // ---- tactic_q = h @ fc2_w^T + b  (waves 0..3, one 16x16 tile each) -------
  if (wave < 4) {
    int tr = wave * 16;
    v8f aq = z8();
    for (int kb = 0; kb < 8; ++kb) {
      FragB a  = frag_lds_rowmajor(R4, HD, tr, kb * 32, lane);
      FragB bb = frag_w_sw(w_fc2, 8, 0, kb, lane);
      aq = wmma_bf16(a, bb, aq);
    }
    int n = lane & 15, hi = lane >> 4;
    float* tqOut = out + (size_t)b * NAG * TN;
#pragma unroll
    for (int v = 0; v < 8; ++v) {
      int row = tr + v + hi * 8;
      tqOut[row * TN + n] = aq[v] + fc2_b[n];
    }
  }

  // ---- pass-through drop mask as float -------------------------------------
  {
    float* dOut = out + TQ + HT + (size_t)b * NAG * NAG;
    for (int i = tid; i < NAG * NAG; i += BDIM)
      dOut[i] = dB[i] ? 1.0f : 0.0f;
  }
}

// ---------------------------------------------------------------------------
// Host launcher.
// ---------------------------------------------------------------------------
extern "C" void kernel_launch(void* const* d_in, const int* in_sizes, int n_in,
                              void* d_out, int out_size, void* d_ws, size_t ws_size,
                              hipStream_t stream) {
  const float*         inputs = (const float*)d_in[0];
  const float*         hidden = (const float*)d_in[1];
  const unsigned char* drop   = (const unsigned char*)d_in[2];
  // d_in[3] == t (unused)
  const float* fc_gru_w   = (const float*)d_in[4];
  const float* fc_gru_b   = (const float*)d_in[5];
  const float* in_proj_w  = (const float*)d_in[6];
  const float* in_proj_b  = (const float*)d_in[7];
  const float* out_proj_w = (const float*)d_in[8];
  const float* out_proj_b = (const float*)d_in[9];
  const float* att_w      = (const float*)d_in[10];
  const float* att_b      = (const float*)d_in[11];
  const float* gru_wih    = (const float*)d_in[12];
  const float* gru_whh    = (const float*)d_in[13];
  const float* gru_bih    = (const float*)d_in[14];
  const float* gru_bhh    = (const float*)d_in[15];
  const float* fc2_w      = (const float*)d_in[16];
  const float* fc2_b      = (const float*)d_in[17];

  // d_ws layout (dwords): swizzled bf16 weights, ~1.7 MB total (L2-resident).
  unsigned* ws = (unsigned*)d_ws;
  unsigned* w_fc   = ws;                 // 256x256 -> 32768 dwords
  unsigned* w_inp  = ws + 32768;         // 768x256 -> 98304
  unsigned* w_outp = ws + 131072;        // 256x256 -> 32768
  unsigned* w_att  = ws + 163840;        // 256x512 -> 65536
  unsigned* w_ih   = ws + 229376;        // 768x256 -> 98304
  unsigned* w_hh   = ws + 327680;        // 768x256 -> 98304
  unsigned* w_fc2  = ws + 425984;        // 16x256  -> 2048

  auto blocks = [](int dwords) { return (dwords + BDIM - 1) / BDIM; };
  swizzle_w<<<blocks(32768), BDIM, 0, stream>>>(fc_gru_w,   256, 256, w_fc);
  swizzle_w<<<blocks(98304), BDIM, 0, stream>>>(in_proj_w,  768, 256, w_inp);
  swizzle_w<<<blocks(32768), BDIM, 0, stream>>>(out_proj_w, 256, 256, w_outp);
  swizzle_w<<<blocks(65536), BDIM, 0, stream>>>(att_w,      256, 512, w_att);
  swizzle_w<<<blocks(98304), BDIM, 0, stream>>>(gru_wih,    768, 256, w_ih);
  swizzle_w<<<blocks(98304), BDIM, 0, stream>>>(gru_whh,    768, 256, w_hh);
  swizzle_w<<<blocks(2048),  BDIM, 0, stream>>>(fc2_w,       16, 256, w_fc2);

  const int Bdim = in_sizes[0] / (NAG * DIN);              // 2048
  const size_t shmem = 110592u * sizeof(unsigned short);   // 216 KB < 320 KB/WGP

  tactic_fused<<<Bdim, BDIM, shmem, stream>>>(
      inputs, hidden, drop,
      w_fc, fc_gru_b, w_inp, in_proj_b, w_outp, out_proj_b,
      w_att, att_b, w_ih, w_hh, gru_bih, gru_bhh, w_fc2, fc2_b,
      (float*)d_out);
}
